// LSTMPPunctuator_20040317403347
// MI455X (gfx1250) — matrix operand order repaired and checked by reading, not compiled, not run on hardware
//
#include <hip/hip_runtime.h>
#include <hip/hip_bf16.h>

// ---------- types for WMMA ----------
typedef __attribute__((ext_vector_type(16))) __bf16 v16bf;
typedef __attribute__((ext_vector_type(8)))  __bf16 v8bf;
typedef __attribute__((ext_vector_type(8)))  float  v8f;

// async global->LDS copy types (builtin takes v4i pointers in AS1 / AS3)
typedef __attribute__((ext_vector_type(4))) int v4i;
typedef __attribute__((address_space(1))) v4i* gp_v4i;
typedef __attribute__((address_space(3))) v4i* lp_v4i;

#if __has_builtin(__builtin_amdgcn_global_load_async_to_lds_b128)
#define HAVE_ASYNC_LDS 1
#endif

#if defined(HAVE_ASYNC_LDS)
#if __has_builtin(__builtin_amdgcn_s_wait_asynccnt)
#define WAIT_ASYNC0() __builtin_amdgcn_s_wait_asynccnt(0)
#else
#define WAIT_ASYNC0() asm volatile("s_wait_asynccnt 0x0" ::: "memory")
#endif
#endif

static __device__ __forceinline__ __bf16 f2bf(float f) {
    unsigned u = __builtin_bit_cast(unsigned, f);
    unsigned r = (u + 0x7FFFu + ((u >> 16) & 1u)) >> 16;
    unsigned short s = (unsigned short)r;
    return __builtin_bit_cast(__bf16, s);
}

static __device__ __forceinline__ float sigm(float x) {
    return 1.0f / (1.0f + __expf(-x));
}

// Load A fragment (16x32 bf16, MxK): rowptr points at (row base + k + 16*lhi)
static __device__ __forceinline__ v16bf load_afrag(const __bf16* rowptr) {
    v8bf a0 = *(const v8bf*)(rowptr);
    v8bf a1 = *(const v8bf*)(rowptr + 8);
    return __builtin_shufflevector(a0, a1, 0,1,2,3,4,5,6,7,8,9,10,11,12,13,14,15);
}

#define NWG 64  // persistent workgroups (one per WGP; all resident)

// device-wide barrier: monotonic generation counter, agent scope
static __device__ __forceinline__ void grid_barrier(int* bar, int gen) {
    __threadfence();
    __syncthreads();
    if (threadIdx.x == 0) {
        __hip_atomic_fetch_add(bar, 1, __ATOMIC_RELEASE, __HIP_MEMORY_SCOPE_AGENT);
        while (__hip_atomic_load(bar, __ATOMIC_ACQUIRE, __HIP_MEMORY_SCOPE_AGENT) < gen * NWG) {
            __builtin_amdgcn_s_sleep(1);
        }
    }
    __syncthreads();
    __threadfence();
}

// stage this WG's 64 xg columns for timestep t into an LDS buffer (16KB)
// layout: float[64 n][64 cols] with col = gate*16 + hl
static __device__ __forceinline__ void stage_xg(const float* __restrict__ xg,
                                                int t, int h0, char* buf) {
#pragma unroll
    for (int j = 0; j < 4; j++) {
        int c   = threadIdx.x + 256 * j;  // 1024 chunks of 16B
        int seg = c >> 2;                 // n*4 + gate
        int sub = c & 3;
        int n = seg >> 2, gate = seg & 3;
        const float* gsrc = xg + ((size_t)n * 512 + t) * 4096 + gate * 1024 + h0 + sub * 4;
        char* dst = buf + ((size_t)n * 64 + gate * 16 + sub * 4) * 4;
#if defined(HAVE_ASYNC_LDS)
        __builtin_amdgcn_global_load_async_to_lds_b128(
            (gp_v4i)(unsigned long long)gsrc,
            (lp_v4i)(unsigned int)(unsigned long long)dst, 0, 0);
#else
        *(float4*)dst = *(const float4*)gsrc;
#endif
    }
}

// ---------- elementwise helpers ----------
__global__ void convert_f32_bf16_kernel(const float* __restrict__ src,
                                        __bf16* __restrict__ dst, long n) {
    long i = (long)blockIdx.x * blockDim.x + threadIdx.x;
    long stride = (long)gridDim.x * blockDim.x;
    for (; i < n; i += stride) dst[i] = f2bf(src[i]);
}

__global__ void embed_kernel(const int* __restrict__ tokens,
                             const float* __restrict__ emb,
                             __bf16* __restrict__ xbf) {
    const int E = 512;
    long i = (long)blockIdx.x * blockDim.x + threadIdx.x; // over N*T*E
    long nt = i / E;
    int  e  = (int)(i % E);
    int  tok = tokens[nt];
    xbf[i] = f2bf(emb[(size_t)tok * E + e]);
}

__global__ void init_state_kernel(const float* __restrict__ h0l,
                                  __bf16* __restrict__ rbf,
                                  int* __restrict__ bar) {
    int i = blockIdx.x * blockDim.x + threadIdx.x; // over N*P = 32768
    if (i == 0) *bar = 0;
    if (i < 64 * 512) rbf[i] = f2bf(h0l[i]);
}

// ---------- big input GEMM: xg = x(bf16) @ Wx^T + b ----------
__global__ void gemm_xg_kernel(const __bf16* __restrict__ A,
                               const __bf16* __restrict__ B,
                               const float* __restrict__ bias,
                               float* __restrict__ Cout) {
    const int K = 512, Nc = 4096;
    int gtid = blockIdx.x * blockDim.x + threadIdx.x;
    int wave = gtid >> 5;
    int lane = threadIdx.x & 31;
    int lan  = lane & 15;
    int lhi  = lane >> 4;
    int ntile  = wave & 255;  // 256 N tiles of 16
    int mgroup = wave >> 8;   // 512 M groups of 64
    int col = ntile * 16 + lan;
    int m0  = mgroup * 64;

    float bval = bias[col];
    v8f acc[4];
#pragma unroll
    for (int i = 0; i < 4; i++)
#pragma unroll
        for (int v = 0; v < 8; v++) acc[i][v] = bval;

    const __bf16* Brow = B + (size_t)col * K;
    for (int k = 0; k < K; k += 32) {
        v16bf bfrag = *(const v16bf*)(Brow + k + 16 * lhi);
#pragma unroll
        for (int i = 0; i < 4; i++) {
            const __bf16* Arow = A + (size_t)(m0 + i * 16 + lan) * K + k + 16 * lhi;
            v16bf afrag = load_afrag(Arow);
            acc[i] = __builtin_amdgcn_wmma_f32_16x16x32_bf16(
                false, afrag, false, bfrag, (short)0, acc[i], false, false);
        }
    }
#pragma unroll
    for (int i = 0; i < 4; i++) {
#pragma unroll
        for (int v = 0; v < 8; v++) {
            int row = m0 + i * 16 + v + 8 * lhi;
            Cout[(size_t)row * Nc + col] = acc[i][v];
        }
    }
}

// ---------- persistent LSTMP layer kernel ----------
// grid: NWG=64 blocks x 256 threads. Block wg owns H chunk [wg*16, wg*16+16).
// Dyn LDS (152KB): Wr 64K | Wp 32K | xg double-buffer 2x16K | gates 16K | reduce 8K
__global__ void lstm_layer_persistent(
    const float*  __restrict__ xg,    // (N,T,4H)
    const __bf16* __restrict__ Wr,    // (4H,P) layer slice
    const __bf16* __restrict__ Wp,    // (P,H)  layer slice
    const float*  __restrict__ peep,  // (3,H)
    const float*  __restrict__ c0l,   // (N,H)
    float*        __restrict__ out_seq, // (N,T,P)
    __bf16*       __restrict__ rbf,   // (N,P)  recurrent state (bf16)
    __bf16*       __restrict__ sbf,   // (N,H)  o*tanh(c) staging (bf16)
    float*        __restrict__ hs_out,  // (N,P)
    float*        __restrict__ cs_out,  // (N,H)
    int*          __restrict__ bar) {
    const int H = 1024, P = 512, T = 512;
    extern __shared__ char smem[];
    __bf16* WrS  = (__bf16*)(smem);            // [64][512]   row = gate*16+hl
    __bf16* WpS  = (__bf16*)(smem + 65536);    // [16][1024]  row = output-col lane
    char*   xgSb = smem + 98304;               // 2 x 16KB staging buffers
    float*  xgSf = (float*)(smem + 98304);     // float view [buf][64 n][64 col]
    float*  gS   = (float*)(smem + 131072);    // [4][64][16]
    float*  red  = (float*)(smem + 147456);    // [2][4][32][8]

    const int wg   = blockIdx.x;       // 0..63
    const int h0   = wg * 16;
    const int lane = threadIdx.x & 31;
    const int wave = threadIdx.x >> 5; // 0..7
    const int lan  = lane & 15;
    const int lhi  = lane >> 4;

    // ---- stage Wr slice (4 gates x 16 cols x 512 K) into LDS ----
    for (int i = threadIdx.x; i < 4096; i += 256) {   // 4096 chunks of 8 bf16
        int row  = i >> 6;            // 0..63 (gate*16+hl)
        int ck   = (i & 63) * 8;
        int gate = row >> 4, hl = row & 15;
        *(v8bf*)(WrS + (size_t)row * 512 + ck) =
            *(const v8bf*)(Wr + (size_t)(gate * H + h0 + hl) * P + ck);
    }
    // ---- stage Wp slice (16 cols x 1024 K) into LDS ----
    const int ntp = wg >> 1;          // proj N-tile for this WG
    for (int i = threadIdx.x; i < 2048; i += 256) {   // 2048 chunks of 8 bf16
        int row = i >> 7;             // 0..15
        int ck  = (i & 127) * 8;
        *(v8bf*)(WpS + (size_t)row * 1024 + ck) =
            *(const v8bf*)(Wp + (size_t)(ntp * 16 + row) * H + ck);
    }

#if defined(HAVE_ASYNC_LDS)
    stage_xg(xg, 0, h0, xgSb);        // prefetch t=0 into buffer 0
#endif

    // ---- cell state in registers: thread handles (n, hl=tid&15), n = tid>>4 + 16j
    const int hl_t = threadIdx.x & 15;
    const int h_t  = h0 + hl_t;
    float pci = peep[0 * H + h_t], pcf = peep[1 * H + h_t], pco = peep[2 * H + h_t];
    float c_reg[4];
#pragma unroll
    for (int j = 0; j < 4; j++) {
        int n = (threadIdx.x >> 4) + 16 * j;
        c_reg[j] = c0l[n * H + h_t];
    }
    __syncthreads();

    // proj assignment: tile_local = wave>>2 (0/1), K-split wsub = wave&3
    const int tlp  = wave >> 2;
    const int wsub = wave & 3;
    const int mtp  = (2 * wg + tlp) & 3;

    int gen = 0;
    for (int t = 0; t < T; t++) {
#if defined(HAVE_ASYNC_LDS)
        WAIT_ASYNC0();                 // xg slice for step t landed in LDS
        __syncthreads();
        const float* xgT = xgSf + (size_t)(t & 1) * 4096;
#else
        stage_xg(xg, t, h0, xgSb);     // synchronous fallback, single buffer
        __syncthreads();
        const float* xgT = xgSf;
#endif

        // ===== phase 1: gates g = xg[:,t,:] + r @ Wr^T (WMMA, B from LDS) =====
#pragma unroll
        for (int i = 0; i < 2; i++) {
            int tile = wave * 2 + i;   // 0..15
            int mt   = tile & 3;
            int gate = tile >> 2;

            v8f acc;
#pragma unroll
            for (int v = 0; v < 8; v++) {
                int n = mt * 16 + v + 8 * lhi;
                acc[v] = xgT[(size_t)n * 64 + gate * 16 + lan];
            }
            const __bf16* BrowL = WrS + (size_t)(gate * 16 + lan) * 512;
            for (int k = 0; k < P; k += 32) {
                v16bf bfrag = *(const v16bf*)(BrowL + k + 16 * lhi);
                const __bf16* Arow = rbf + (size_t)(mt * 16 + lan) * P + k + 16 * lhi;
                v16bf afrag = load_afrag(Arow);
                acc = __builtin_amdgcn_wmma_f32_16x16x32_bf16(
                    false, afrag, false, bfrag, (short)0, acc, false, false);
            }
#pragma unroll
            for (int v = 0; v < 8; v++) {
                int n = mt * 16 + v + 8 * lhi;
                gS[((size_t)gate * 64 + n) * 16 + lan] = acc[v];
            }
        }
        __syncthreads();

#if defined(HAVE_ASYNC_LDS)
        // prefetch next step's xg slice while we do nonlinearity/proj/barriers
        if (t + 1 < T) stage_xg(xg, t + 1, h0, xgSb + ((t + 1) & 1) * 16384);
#endif

        // ===== phase 2: peephole nonlinearity, cell update, s = o*tanh(c) =====
#pragma unroll
        for (int j = 0; j < 4; j++) {
            int n = (threadIdx.x >> 4) + 16 * j;
            float gi = gS[((size_t)0 * 64 + n) * 16 + hl_t];
            float gf = gS[((size_t)1 * 64 + n) * 16 + hl_t];
            float gg = gS[((size_t)2 * 64 + n) * 16 + hl_t];
            float go = gS[((size_t)3 * 64 + n) * 16 + hl_t];
            float cold = c_reg[j];
            float iv = sigm(gi + pci * cold);
            float fv = sigm(gf + pcf * cold);
            float cn = fv * cold + iv * tanhf(gg);
            float ov = sigm(go + pco * cn);
            c_reg[j] = cn;
            sbf[n * H + h_t] = f2bf(ov * tanhf(cn));
        }
        grid_barrier(bar, ++gen);   // s visible device-wide

        // ===== phase 3: projection r = s @ Wp^T (K=1024 split over 4 waves) =====
        {
            v8f acc;
#pragma unroll
            for (int v = 0; v < 8; v++) acc[v] = 0.0f;
            const __bf16* BrowL = WpS + (size_t)lan * 1024;
            int kbase = wsub * 256;
            for (int kk = 0; kk < 256; kk += 32) {
                int k = kbase + kk;
                v16bf bfrag = *(const v16bf*)(BrowL + k + 16 * lhi);
                const __bf16* Arow = sbf + (size_t)(mtp * 16 + lan) * H + k + 16 * lhi;
                v16bf afrag = load_afrag(Arow);
                acc = __builtin_amdgcn_wmma_f32_16x16x32_bf16(
                    false, afrag, false, bfrag, (short)0, acc, false, false);
            }
#pragma unroll
            for (int v = 0; v < 8; v++)
                red[(((size_t)tlp * 4 + wsub) * 32 + lane) * 8 + v] = acc[v];
        }
        __syncthreads();
#pragma unroll
        for (int q = 0; q < 2; q++) {
            int e   = threadIdx.x + 256 * q;  // 0..511
            int tl  = e >> 8;
            int rem = e & 255;
            int ln  = rem >> 3;
            int v   = rem & 7;
            float sum = 0.0f;
#pragma unroll
            for (int w = 0; w < 4; w++)
                sum += red[(((size_t)tl * 4 + w) * 32 + ln) * 8 + v];
            int mt  = (2 * wg + tl) & 3;
            int n   = mt * 16 + v + 8 * (ln >> 4);
            int col = ntp * 16 + (ln & 15);
            out_seq[(size_t)n * T * P + (size_t)t * P + col] = sum;
            rbf[(size_t)n * P + col] = f2bf(sum);
            if (t == T - 1) hs_out[(size_t)n * P + col] = sum;
        }
        grid_barrier(bar, ++gen);   // r visible device-wide
    }

    // ---- final cell state ----
#pragma unroll
    for (int j = 0; j < 4; j++) {
        int n = (threadIdx.x >> 4) + 16 * j;
        cs_out[n * H + h_t] = c_reg[j];
    }
}

// ---------- FC head: score = out1 @ fcW^T + fcb (C=5) ----------
__global__ void fc_kernel(const float* __restrict__ x,  // (N*T, P)
                          const float* __restrict__ W,  // (C,P)
                          const float* __restrict__ b,  // (C)
                          float* __restrict__ score) {  // (N*T, C)
    const int P = 512, C = 5;
    __shared__ float Ws[5 * 512];
    for (int i = threadIdx.x; i < C * P; i += blockDim.x) Ws[i] = W[i];
    __syncthreads();
    int m = blockIdx.x * blockDim.x + threadIdx.x; // 0..32767
    const float* xr = x + (size_t)m * P;
    float acc[5];
#pragma unroll
    for (int cc = 0; cc < 5; cc++) acc[cc] = b[cc];
    for (int p = 0; p < P; p++) {
        float xv = xr[p];
#pragma unroll
        for (int cc = 0; cc < 5; cc++) acc[cc] += xv * Ws[cc * P + p];
    }
#pragma unroll
    for (int cc = 0; cc < 5; cc++) score[(size_t)m * C + cc] = acc[cc];
}

extern "C" void kernel_launch(void* const* d_in, const int* in_sizes, int n_in,
                              void* d_out, int out_size, void* d_ws, size_t ws_size,
                              hipStream_t stream) {
    (void)in_sizes; (void)n_in; (void)out_size; (void)ws_size;
    const int E = 512, Hh = 1024, Pp = 512, Nn = 64, Tt = 512, G4 = 4096;

    const int*   tokens = (const int*)d_in[0];
    const float* h0     = (const float*)d_in[1];   // (L,N,P)
    const float* c0     = (const float*)d_in[2];   // (L,N,H)
    const float* emb    = (const float*)d_in[3];   // (V,E)
    const float* Wx     = (const float*)d_in[4];   // (L,4H,E)
    const float* Wr     = (const float*)d_in[5];   // (L,4H,P)
    const float* Wp     = (const float*)d_in[6];   // (L,P,H)
    const float* peep   = (const float*)d_in[7];   // (L,3,H)
    const float* bias   = (const float*)d_in[8];   // (L,4H)
    const float* fcW    = (const float*)d_in[9];   // (C,P)
    const float* fcb    = (const float*)d_in[10];  // (C)

    float* out = (float*)d_out;
    float* score  = out;                      // N*T*C = 163840
    float* hs_out = out + 163840;             // L*N*P = 65536
    float* cs_out = out + 163840 + 65536;     // L*N*H = 131072

    // workspace layout (256B aligned)
    char* Wb = (char*)d_ws;
    size_t off = 0;
    auto take = [&](size_t bytes) -> char* {
        char* p = Wb + off;
        off += (bytes + 255) & ~(size_t)255;
        return p;
    };
    float*  xg     = (float*) take((size_t)Nn * Tt * G4 * 4);   // 512 MB
    __bf16* xbf    = (__bf16*)take((size_t)Nn * Tt * E * 2);
    float*  out0   = (float*) take((size_t)Nn * Tt * Pp * 4);
    float*  out1   = (float*) take((size_t)Nn * Tt * Pp * 4);
    __bf16* Wx_bf  = (__bf16*)take((size_t)2 * G4 * E * 2);
    __bf16* Wr_bf  = (__bf16*)take((size_t)2 * G4 * Pp * 2);
    __bf16* Wp_bf  = (__bf16*)take((size_t)2 * Pp * Hh * 2);
    __bf16* r_bf   = (__bf16*)take((size_t)Nn * Pp * 2);
    __bf16* s_bf   = (__bf16*)take((size_t)Nn * Hh * 2);
    int*    bar    = (int*)   take(256);

    // weight conversion (bf16)
    convert_f32_bf16_kernel<<<4096, 256, 0, stream>>>(Wx, Wx_bf, (long)2 * G4 * E);
    convert_f32_bf16_kernel<<<4096, 256, 0, stream>>>(Wr, Wr_bf, (long)2 * G4 * Pp);
    convert_f32_bf16_kernel<<<4096, 256, 0, stream>>>(Wp, Wp_bf, (long)2 * Pp * Hh);

    // embedding gather -> bf16 layer-0 input
    embed_kernel<<<(Nn * Tt * E) / 256, 256, 0, stream>>>(tokens, emb, xbf);

    const size_t dynLds = 65536 + 32768 + 32768 + 16384 + 8192;  // 152 KB

    for (int l = 0; l < 2; l++) {
        float* out_l = (l == 0) ? out0 : out1;

        gemm_xg_kernel<<<16384, 256, 0, stream>>>(
            xbf, Wx_bf + (size_t)l * G4 * E, bias + (size_t)l * G4, xg);

        init_state_kernel<<<128, 256, 0, stream>>>(h0 + (size_t)l * Nn * Pp, r_bf, bar);

        lstm_layer_persistent<<<NWG, 256, dynLds, stream>>>(
            xg, Wr_bf + (size_t)l * G4 * Pp, Wp_bf + (size_t)l * Pp * Hh,
            peep + (size_t)l * 3 * Hh, c0 + (size_t)l * Nn * Hh,
            out_l, r_bf, s_bf,
            hs_out + (size_t)l * Nn * Pp, cs_out + (size_t)l * Nn * Hh, bar);

        if (l == 0) {
            convert_f32_bf16_kernel<<<4096, 256, 0, stream>>>(
                out0, xbf, (long)Nn * Tt * Pp);
        }
    }

    fc_kernel<<<128, 256, 0, stream>>>(out1, fcW, fcb, score);
}
